// hybrid_neuron_9904194585304
// MI455X (gfx1250) — matrix operand order repaired and checked by reading, not compiled, not run on hardware
//
#include <hip/hip_runtime.h>
#include <math.h>

// MI455X / gfx1250: wave32, WMMA bf16 16x16x32 (f32 accumulate).
//
// Persistent-strip design: the P-recurrence is separable across OUT rows, so
// each workgroup owns a 16-row strip and runs all T=64 steps with only
// workgroup barriers (one launch, no global sync). Recurrent state (P, S as
// bf16 WMMA A-operands) lives in LDS; the f32 P state and the mlp strip live
// in registers (ownership matches the WMMA C-fragment layout); x/x^T stream
// from L2 as coalesced 16B loads.
//
// Register tuning (from compile feedback): 512 threads = 16 waves, each wave
// owns 2 16x16 tiles (32 cols). 4 waves/SIMD -> ~128 VGPR budget; persistent
// state is 6 x v8f (48 VGPRs) + ~3 fragments. K-loops are limited to
// "#pragma unroll 2" so the compiler keeps a 2-deep load/WMMA pipeline
// without hoisting all 16 k-steps (which caused >256-VGPR allocation,
// s_set_vgpr_msb churn, and scratch spills with 4 tiles/wave).
#define DIM 512            // B == IN == OUT
#define NSTEPS 64
#define STRIP 16           // OUT rows per workgroup -> 32 workgroups
#define THREADS 512        // 16 waves
#define TILES 2            // 16-col tiles per wave: 16*2*16 = 512 cols

typedef __attribute__((ext_vector_type(16))) __bf16 v16bf;
typedef __attribute__((ext_vector_type(8)))  __bf16 v8bf;
typedef __attribute__((ext_vector_type(8)))  float  v8f;

union V16U { v16bf v; v8bf h[2]; };

// A fragment (16x32 bf16, M x K) from row-major [16, DIM] strip.
// ISA layout: lane L(0-15)=row M; VGPR0-3 hold K=kb..kb+7, VGPR4-7 K=kb+16..kb+23,
// kb = kk + (lane>=16 ? 8 : 0). Two contiguous 16B loads per lane.
__device__ __forceinline__ v16bf a_frag(const __bf16* A, int kk, int lane) {
  int kb = kk + ((lane >> 4) << 3);
  const __bf16* p = A + (lane & 15) * DIM + kb;
  V16U u;
  u.h[0] = *(const v8bf*)(p);
  u.h[1] = *(const v8bf*)(p + 16);
  return u.v;
}

// B fragment (32x16 bf16, K x N) from BT stored row-major [N, DIM].
// Lane L(0-15)=col N holds K=kk..kk+7, kk+16..kk+23; lanes 16-31 hold K=kk+8..kk+15, kk+24..kk+31.
__device__ __forceinline__ v16bf bt_frag(const __bf16* BT, int col0, int kk, int lane) {
  int kb = kk + ((lane >> 4) << 3);
  const __bf16* p = BT + (col0 + (lane & 15)) * DIM + kb;
  V16U u;
  u.h[0] = *(const v8bf*)(p);
  u.h[1] = *(const v8bf*)(p + 16);
  return u.v;
}

__device__ __forceinline__ v8f wmma_bf16(v16bf a, v16bf b, v8f c) {
  return __builtin_amdgcn_wmma_f32_16x16x32_bf16(false, a, false, b, (short)0, c,
                                                 false, false);
}

// ---------------- setup: f32 -> bf16 conversions (L2-resident operands) ----
__global__ __launch_bounds__(256) void cvt_x_kernel(const float* __restrict__ x,
                                                    __bf16* __restrict__ xb,
                                                    __bf16* __restrict__ xTb) {
  int idx = blockIdx.x * 256 + threadIdx.x;      // < DIM*DIM
  int b = idx >> 9, i = idx & (DIM - 1);
  float v = x[idx];
  xb[idx] = (__bf16)v;                           // x   [B, IN]
  xTb[i * DIM + b] = (__bf16)v;                  // x^T [IN, B]
}

__global__ __launch_bounds__(256) void cvt_w_kernel(const float* __restrict__ W,
                                                    __bf16* __restrict__ Wb) {
  int idx = blockIdx.x * 256 + threadIdx.x;
  Wb[idx] = (__bf16)W[idx];
}

// ---------------- persistent strip kernel: whole T loop, no global sync ----
__global__ __launch_bounds__(THREADS) void snn_strip_kernel(
    const float* __restrict__ P0, const __bf16* __restrict__ Wb,
    const __bf16* __restrict__ xb, const __bf16* __restrict__ xTb,
    const float* __restrict__ bias, const float* __restrict__ alpha,
    const float* __restrict__ eta, const float* __restrict__ beta,
    float* __restrict__ out) {
  __shared__ __bf16 sP[STRIP * DIM];             // P strip, bf16 (WMMA A operand)
  __shared__ __bf16 sS[STRIP * DIM];             // H(u)+beta strip, bf16

  const int row0 = blockIdx.x * STRIP;
  const int wave = threadIdx.x >> 5;
  const int lane = threadIdx.x & 31;
  const int hi = lane >> 4, lc = lane & 15;
  const int col0 = wave * (16 * TILES);          // this wave's 32-column slice

  const float al = alpha[0], et = eta[0], be = beta[0];
  const float kc = 0.2f;                         // dt / tao_u
  const float Vth = 0.2f;
  const float decay = 0.36787944117144233f;      // exp(-dt/tao_w)

  // ---- init: P0 strip -> LDS bf16 mirror + per-lane f32 registers ----
  for (int i = threadIdx.x; i < STRIP * DIM; i += THREADS)
    sP[i] = (__bf16)P0[row0 * DIM + i];
  v8f pacc[TILES];                               // P f32 state (C-layout ownership)
#pragma unroll
  for (int j = 0; j < TILES; ++j)
#pragma unroll
    for (int v = 0; v < 8; ++v)
      pacc[j][v] = P0[(row0 + hi * 8 + v) * DIM + col0 + j * 16 + lc];

  // ---- mlp strip: relu(W_strip @ x^T + b), held in registers ----
  v8f mlp[TILES] = {};
  const __bf16* Wstrip = Wb + row0 * DIM;
#pragma unroll 2
  for (int kk = 0; kk < DIM; kk += 32) {
    v16bf a = a_frag(Wstrip, kk, lane);
#pragma unroll
    for (int j = 0; j < TILES; ++j)
      mlp[j] = wmma_bf16(a, bt_frag(xb, col0 + j * 16, kk, lane), mlp[j]);
  }
  {
    const v8f bv = *(const v8f*)(bias + row0 + hi * 8);
#pragma unroll
    for (int j = 0; j < TILES; ++j)
#pragma unroll
      for (int v = 0; v < 8; ++v) {
        float r = mlp[j][v] + bv[v];
        mlp[j][v] = r > 0.f ? r : 0.f;
      }
  }
  __syncthreads();

  for (int t = 0; t < NSTEPS; ++t) {
    const float e_t = __expf(-(float)t);

    // GEMM1: acc = (P @ x^T) tiles. A = sP (LDS ds_load), BT = x (global, L2).
    v8f acc[TILES] = {};
#pragma unroll 2
    for (int kk = 0; kk < DIM; kk += 32) {
      v16bf a = a_frag(sP, kk, lane);
#pragma unroll
      for (int j = 0; j < TILES; ++j)
        acc[j] = wmma_bf16(a, bt_frag(xb, col0 + j * 16, kk, lane), acc[j]);
    }
    // u = k*(mlp*e_t + alpha*acc); spikes -> HBM; S = H(u)+beta -> LDS.
    float* outt = out + (size_t)t * DIM * DIM;   // [B, OUT] slice
#pragma unroll
    for (int j = 0; j < TILES; ++j) {
#pragma unroll
      for (int v = 0; v < 8; ++v) {
        int m = hi * 8 + v;                      // strip-local OUT row
        int n = col0 + j * 16 + lc;              // B column
        float u = kc * (mlp[j][v] * e_t + al * acc[j][v]);
        outt[n * DIM + row0 + m] = (u > Vth) ? 1.f : 0.f;  // 8 consecutive floats/lane
        sS[m * DIM + n] = (__bf16)(((u > 0.f) ? 1.f : 0.f) + be);
      }
    }
    __syncthreads();                             // S strip complete, P reads done

    // GEMM2: acc2 = (S @ x) tiles. A = sS (LDS), BT = x^T (global, L2).
    v8f acc2[TILES] = {};
#pragma unroll 2
    for (int kk = 0; kk < DIM; kk += 32) {
      v16bf a = a_frag(sS, kk, lane);
#pragma unroll
      for (int j = 0; j < TILES; ++j)
        acc2[j] = wmma_bf16(a, bt_frag(xTb, col0 + j * 16, kk, lane), acc2[j]);
    }
#pragma unroll
    for (int j = 0; j < TILES; ++j) {
#pragma unroll
      for (int v = 0; v < 8; ++v) {
        float p = pacc[j][v] * decay + et * acc2[j][v];  // f32 recurrence in regs
        pacc[j][v] = p;
        sP[(hi * 8 + v) * DIM + col0 + j * 16 + lc] = (__bf16)p;
      }
    }
    __syncthreads();                             // P strip ready for next step
  }
}

extern "C" void kernel_launch(void* const* d_in, const int* in_sizes, int n_in,
                              void* d_out, int out_size, void* d_ws, size_t ws_size,
                              hipStream_t stream) {
  const float* x     = (const float*)d_in[0];   // [B, IN]
  const float* W     = (const float*)d_in[1];   // [OUT, IN]
  const float* bias  = (const float*)d_in[2];   // [OUT]
  const float* alpha = (const float*)d_in[3];
  const float* eta   = (const float*)d_in[4];
  const float* beta  = (const float*)d_in[5];
  const float* P0    = (const float*)d_in[6];   // [OUT, IN]
  (void)in_sizes; (void)n_in; (void)out_size; (void)ws_size;

  char* ws = (char*)d_ws;                       // 1.5 MB used, L2-resident
  __bf16* xb  = (__bf16*)(ws + 0x000000);       // x   bf16 [B, IN]   512 KB
  __bf16* xTb = (__bf16*)(ws + 0x080000);       // x^T bf16 [IN, B]   512 KB
  __bf16* Wb  = (__bf16*)(ws + 0x100000);       // W   bf16 [OUT, IN] 512 KB
  float*  out = (float*)d_out;                  // [T, B, OUT]

  const int elem_blocks = (DIM * DIM) / 256;    // 1024
  cvt_x_kernel<<<elem_blocks, 256, 0, stream>>>(x, xb, xTb);
  cvt_w_kernel<<<elem_blocks, 256, 0, stream>>>(W, Wb);

  // One persistent kernel runs the entire 64-step recurrence: each of the 32
  // workgroups owns an independent 16-row strip of P/u/S (row-separable).
  snn_strip_kernel<<<DIM / STRIP, THREADS, 0, stream>>>(P0, Wb, xb, xTb, bias,
                                                        alpha, eta, beta, out);
}
